// SNNSLiCell_60902636257671
// MI455X (gfx1250) — compile-verified, matrix operand order checked by reading, not run on hardware
//
#include <hip/hip_runtime.h>
#include <hip/hip_bf16.h>

// ---------------------------------------------------------------------------
// SNN SLi cell for MI455X (gfx1250, wave32, WMMA).
//   pre:    xbf = bf16(x); WiT/WrT/WoT = bf16 transposed weights
//   stage1: cur_in = xbf @ Wi + bi          (bf16 WMMA GEMM, 32768x2048x2048)
//   stage2: 512 serialized LIF steps:  cur = v @ Wr + br ; pointwise update
//   stage3: out = spikes @ Wo + bo          (bf16 WMMA GEMM, 32768x2048x2048)
// ---------------------------------------------------------------------------

typedef __bf16 bf16;
typedef __attribute__((ext_vector_type(16))) __bf16 v16bf;
typedef __attribute__((ext_vector_type(8)))  __bf16 v8bf;
typedef __attribute__((ext_vector_type(8)))  float  v8f;
typedef __attribute__((ext_vector_type(4)))  float  v4f;

#define BB 64
#define SS 512
#define HH 2048
#define BS (BB * SS)          // 32768 rows in the big GEMMs
#define VN (BB * HH)          // 131072 state elements

// ---------------- workspace layout (bytes) ----------------
#define OFF_WIT   ((size_t)0)                        // H*H bf16  = 8 MB
#define OFF_WRT   (OFF_WIT + (size_t)HH*HH*2)
#define OFF_WOT   (OFF_WRT + (size_t)HH*HH*2)
#define OFF_CURIN (OFF_WOT + (size_t)HH*HH*2)        // B*S*H f32  = 256 MB
#define OFF_SPK   (OFF_CURIN + (size_t)BS*HH*4)      // B*S*H bf16 = 128 MB
#define OFF_XBF   (OFF_SPK + (size_t)BS*HH*2)        // B*S*H bf16 = 128 MB
#define OFF_V     (OFF_XBF + (size_t)BS*HH*2)        // B*H f32
#define OFF_THR   (OFF_V   + (size_t)VN*4)
#define OFF_VBF   (OFF_THR + (size_t)VN*4)           // B*H bf16
#define OFF_CUR   (OFF_VBF + (size_t)VN*2)           // B*H f32

// ---------------------------------------------------------------------------
// Bulk f32 -> bf16 convert (8 elements / thread, b128 in, b128 out)
// ---------------------------------------------------------------------------
__global__ __launch_bounds__(256) void f32_to_bf16_vec(
    const float* __restrict__ in, bf16* __restrict__ out)
{
    size_t i = ((size_t)blockIdx.x * 256 + threadIdx.x) * 8;
    v4f f0 = *(const v4f*)(in + i);
    v4f f1 = *(const v4f*)(in + i + 4);
    v8bf o;
#pragma unroll
    for (int j = 0; j < 4; ++j) { o[j] = (bf16)f0[j]; o[4 + j] = (bf16)f1[j]; }
    *(v8bf*)(out + i) = o;
}

// ---------------------------------------------------------------------------
// Transpose + f32->bf16 downconvert:  WT[n][k] = (bf16)W[k][n]
// ---------------------------------------------------------------------------
__global__ __launch_bounds__(256) void transpose_to_bf16(
    const float* __restrict__ W, bf16* __restrict__ WT)
{
    __shared__ float tile[32][33];
    int x  = blockIdx.x * 32 + threadIdx.x;   // col of W
    int y0 = blockIdx.y * 32 + threadIdx.y;   // row of W (ty in [0,8))
#pragma unroll
    for (int i = 0; i < 32; i += 8)
        tile[threadIdx.y + i][threadIdx.x] = W[(size_t)(y0 + i) * HH + x];
    __syncthreads();
    int x2 = blockIdx.y * 32 + threadIdx.x;   // col of WT (= row of W)
    int y2 = blockIdx.x * 32 + threadIdx.y;   // row of WT (= col of W)
#pragma unroll
    for (int i = 0; i < 32; i += 8)
        WT[(size_t)(y2 + i) * HH + x2] = (bf16)tile[threadIdx.x][threadIdx.y + i];
}

// ---------------------------------------------------------------------------
// bf16 A/B fragment load: two contiguous 16B chunks at k and k+16
// (16-bit 16x32 WMMA layout: lane l -> row/col l%16, kbase (l/16)*8)
// ---------------------------------------------------------------------------
__device__ __forceinline__ v16bf load_frag(const bf16* __restrict__ p)
{
    v8bf lo = *(const v8bf*)(p);
    v8bf hi = *(const v8bf*)(p + 16);
    v16bf r;
#pragma unroll
    for (int i = 0; i < 8; ++i) { r[i] = lo[i]; r[8 + i] = hi[i]; }
    return r;
}

// ---------------------------------------------------------------------------
// GEMM: C[M][N] = A[M][K] * BT[N][K]^T + bias[N]   (bf16 WMMA, f32 accumulate)
// One wave -> (16*MT) x (16*NT) tile.  Outer loop steps K by 128 with a fully
// unrolled 4x(K=32) body: fragment base pointers are loop-invariant, so all
// loads inside the group use immediate offsets (no per-step 64-bit address
// VALU competing with WMMA issue / triggering hazard NOPs).
// ---------------------------------------------------------------------------
template <int MT, int NT>
__global__ __launch_bounds__(256) void gemm_wmma_bf16(
    const bf16* __restrict__ A, const bf16* __restrict__ BT,
    const float* __restrict__ bias, float* __restrict__ C,
    int M, int N, int K)
{
    const int lane = threadIdx.x & 31;
    const int wave = threadIdx.x >> 5;
    const int wpb  = blockDim.x >> 5;
    const int m0   = (blockIdx.y * wpb + wave) * (16 * MT);
    const int n0   = blockIdx.x * (16 * NT);
    if (m0 >= M) return;

    const int rowb = m0 + (lane & 15);
    const int koff = (lane >> 4) * 8;
    const int ncol = n0 + (lane & 15);

    // loop-invariant fragment base pointers
    const bf16* Ab[MT];
#pragma unroll
    for (int mi = 0; mi < MT; ++mi)
        Ab[mi] = A + (size_t)(rowb + 16 * mi) * K + koff;
    const bf16* Bb[NT];
#pragma unroll
    for (int ni = 0; ni < NT; ++ni)
        Bb[ni] = BT + (size_t)(ncol + 16 * ni) * K + koff;

    const v8f zero = {};
    v8f acc[MT][NT];
#pragma unroll
    for (int mi = 0; mi < MT; ++mi)
#pragma unroll
        for (int ni = 0; ni < NT; ++ni) acc[mi][ni] = zero;

    for (int kb = 0; kb < K; kb += 128) {
#pragma unroll
        for (int kk = 0; kk < 128; kk += 32) {
            v16bf a[MT];
#pragma unroll
            for (int mi = 0; mi < MT; ++mi) {
                if (kk == 0) __builtin_prefetch(Ab[mi] + kb + 256, 0, 0);
                a[mi] = load_frag(Ab[mi] + kb + kk);
            }
            v16bf b[NT];
#pragma unroll
            for (int ni = 0; ni < NT; ++ni) {
                if (kk == 0) __builtin_prefetch(Bb[ni] + kb + 256, 0, 0);
                b[ni] = load_frag(Bb[ni] + kb + kk);
            }
#pragma unroll
            for (int mi = 0; mi < MT; ++mi)
#pragma unroll
                for (int ni = 0; ni < NT; ++ni)
                    acc[mi][ni] = __builtin_amdgcn_wmma_f32_16x16x32_bf16(
                        false, a[mi], false, b[ni], (short)0, acc[mi][ni],
                        false, false);
        }
    }

    // store: VGPR r -> M = tile_m0 + r + 8*(lane/16), N = tile_n0 + (lane%16)
#pragma unroll
    for (int mi = 0; mi < MT; ++mi) {
#pragma unroll
        for (int ni = 0; ni < NT; ++ni) {
            int n = n0 + 16 * ni + (lane & 15);
            float bn = bias[n];
            float* Cp = C + (size_t)(m0 + 16 * mi + ((lane >> 4) * 8)) * N + n;
#pragma unroll
            for (int r = 0; r < 8; ++r)
                Cp[(size_t)r * N] = acc[mi][ni][r] + bn;
        }
    }
}

// ---------------------------------------------------------------------------
// LIF state init: v = 0, thr = -50
// ---------------------------------------------------------------------------
__global__ __launch_bounds__(256) void init_state(
    float* __restrict__ v, float* __restrict__ thr, bf16* __restrict__ vbf)
{
    int i = blockIdx.x * blockDim.x + threadIdx.x;
    if (i < VN) { v[i] = 0.f; thr[i] = -50.f; vbf[i] = (bf16)0.f; }
}

// ---------------------------------------------------------------------------
// Pointwise LIF update for timestep t.
// cur holds v@Wr + br (bias folded into GEMM); cur_in holds x@Wi + bi.
// ---------------------------------------------------------------------------
__global__ __launch_bounds__(256) void lif_update(
    const float* __restrict__ cur_in, const float* __restrict__ cur,
    float* __restrict__ v, float* __restrict__ thr,
    bf16* __restrict__ vbf, bf16* __restrict__ spikes, int t)
{
    int i = blockIdx.x * blockDim.x + threadIdx.x;
    if (i >= VN) return;
    int b = i >> 11;           // /H
    int h = i & (HH - 1);      // %H
    size_t bsidx = ((size_t)b * SS + t) * HH + h;

    float vv = v[i];
    float th = thr[i];
    float current = cur_in[bsidx] + cur[i];

    float dv = (-(vv - (-65.0f)) + current) * (1.0f / 20.0f);
    dv = fminf(fmaxf(dv, -100.0f), 100.0f);
    float nv = vv + dv;
    float sp = (nv >= th) ? 1.0f : 0.0f;
    nv = (sp > 0.0f) ? -70.0f : nv;
    float nth = th + 0.1f * sp - (th - (-50.0f)) * (1.0f / 20.0f);
    nth = fminf(fmaxf(nth, -100.0f), 0.0f);

    v[i]   = nv;
    thr[i] = nth;
    vbf[i] = (bf16)nv;
    spikes[bsidx] = (bf16)sp;
}

// ---------------------------------------------------------------------------
extern "C" void kernel_launch(void* const* d_in, const int* in_sizes, int n_in,
                              void* d_out, int out_size, void* d_ws, size_t ws_size,
                              hipStream_t stream)
{
    const float* x  = (const float*)d_in[0];
    const float* Wi = (const float*)d_in[1];
    const float* bi = (const float*)d_in[2];
    const float* Wr = (const float*)d_in[3];
    const float* br = (const float*)d_in[4];
    const float* Wo = (const float*)d_in[5];
    const float* bo = (const float*)d_in[6];
    float* out = (float*)d_out;

    char* ws = (char*)d_ws;
    bf16*  WiT    = (bf16*)(ws + OFF_WIT);
    bf16*  WrT    = (bf16*)(ws + OFF_WRT);
    bf16*  WoT    = (bf16*)(ws + OFF_WOT);
    float* cur_in = (float*)(ws + OFF_CURIN);
    bf16*  spikes = (bf16*)(ws + OFF_SPK);
    bf16*  xbf    = (bf16*)(ws + OFF_XBF);
    float* v      = (float*)(ws + OFF_V);
    float* thr    = (float*)(ws + OFF_THR);
    bf16*  vbf    = (bf16*)(ws + OFF_VBF);
    float* cur    = (float*)(ws + OFF_CUR);

    // 0) x -> bf16 (one pass; A matrix then fits in 192 MB L2 during stage 1)
    f32_to_bf16_vec<<<(size_t)BS * HH / 8 / 256, 256, 0, stream>>>(x, xbf);

    // 1) weight transpose + bf16 downconvert
    dim3 tb(32, 8), tg(HH / 32, HH / 32);
    transpose_to_bf16<<<tg, tb, 0, stream>>>(Wi, WiT);
    transpose_to_bf16<<<tg, tb, 0, stream>>>(Wr, WrT);
    transpose_to_bf16<<<tg, tb, 0, stream>>>(Wo, WoT);

    // 2) cur_in = x @ Wi + bi    (M=32768, N=2048, K=2048), 32x64 wave tiles
    gemm_wmma_bf16<2, 4><<<dim3(HH / 64, BS / 32 / 8), 256, 0, stream>>>(
        xbf, WiT, bi, cur_in, BS, HH, HH);

    // 3) recurrent LIF scan (latency-bound; maximize wave parallelism)
    init_state<<<(VN + 255) / 256, 256, 0, stream>>>(v, thr, vbf);
    for (int t = 0; t < SS; ++t) {
        // cur = v @ Wr + br   (M=64, N=2048, K=2048) — 256 waves
        gemm_wmma_bf16<1, 2><<<dim3(HH / 32, 1), 128, 0, stream>>>(
            vbf, WrT, br, cur, BB, HH, HH);
        lif_update<<<VN / 256, 256, 0, stream>>>(cur_in, cur, v, thr, vbf, spikes, t);
    }

    // 4) out = spikes @ Wo + bo   (32x64 wave tiles)
    gemm_wmma_bf16<2, 4><<<dim3(HH / 64, BS / 32 / 8), 256, 0, stream>>>(
        spikes, WoT, bo, out, BS, HH, HH);
}
